// Lstmp_14894946583426
// MI455X (gfx1250) — matrix operand order repaired and checked by reading, not compiled, not run on hardware
//
#include <hip/hip_runtime.h>

#define H     50
#define HP    64          // padded hidden
#define GP    256         // padded gate count (4*HP)
#define BT    32          // batch tile per workgroup
#define TT    2048
#define FUT   64
#define OUTW  (TT + FUT)
#define HSTR  72          // h-buffer row stride (halves), 16B-aligned rows
#define GSTR  260         // gates row stride (floats), bank-conflict pad
#define NTHREADS 256

typedef __attribute__((ext_vector_type(16))) _Float16 v16h;
typedef __attribute__((ext_vector_type(8)))  _Float16 v8h;
typedef __attribute__((ext_vector_type(8)))  float    v8f;

__device__ __forceinline__ v8f wmma_f16(v16h a, v16h b, v8f c) {
    // (neg_a, A, neg_b, B, c_mod, C, reuse_a, reuse_b)
    return __builtin_amdgcn_wmma_f32_16x16x32_f16(false, a, false, b, (short)0, c, false, false);
}

// Hardware tanh (TRANS32 op on CDNA5); fallback to libm expansion.
__device__ __forceinline__ float tanh_fast(float x) {
#if __has_builtin(__builtin_amdgcn_tanhf)
    return __builtin_amdgcn_tanhf(x);
#elif __has_builtin(__builtin_amdgcn_tanh_f32)
    return __builtin_amdgcn_tanh_f32(x);
#else
    return tanhf(x);
#endif
}
// sigmoid(x) = 0.5*tanh(x/2) + 0.5  -> single hardware trans op
__device__ __forceinline__ float sigm(float x) {
    return fmaf(0.5f, tanh_fast(0.5f * x), 0.5f);
}

// Build B-operand (32x16 = K x N tile of padded W^T) in WMMA register layout:
// lane 0-15: N = lane, K = kc*32 + {0..7,16..23} ; lane 16-31: N = lane-16, K = kc*32 + {8..15,24..31}.
__device__ __forceinline__ v16h make_B(const float* __restrict__ W, int nt, int kc, int lane) {
    int n_in = lane & 15;
    int gpi  = nt * 16 + n_in;      // padded gate index 0..255
    int g    = gpi >> 6;            // gate block (i,f,g,o)
    int n    = gpi & 63;            // index within padded block
    int kb   = kc * 32 + ((lane >> 4) << 3);
    v16h b;
#pragma unroll
    for (int e = 0; e < 16; ++e) {
        int k = kb + (e < 8 ? e : e + 8);
        float v = (n < H && k < H) ? W[(g * H + n) * H + k] : 0.0f;
        b[e] = (_Float16)v;
    }
    return b;
}

// Load A-operand (16x32 M x K tile of h) from LDS:
// lane 0-15: M=lane, halves 0-7 = K kb..kb+7, halves 8-15 = K kb+16..kb+23 (kb=0)
// lane 16-31: same rows, kb=8.
__device__ __forceinline__ v16h load_A(const _Float16* hs, int mt, int kc, int lane) {
    int row = mt * 16 + (lane & 15);
    int kb  = kc * 32 + ((lane >> 4) << 3);
    const _Float16* p = hs + row * HSTR + kb;
    v8h lo = *(const v8h*)(p);
    v8h hi = *(const v8h*)(p + 16);
    v16h a;
#pragma unroll
    for (int e = 0; e < 8; ++e) { a[e] = lo[e]; a[e + 8] = hi[e]; }
    return a;
}

// Store f32 accumulator tile to gates LDS. D layout: VGPR r, lane l ->
// row = mt*16 + r + 8*(l>>4), col = nt*16 + (l&15).
__device__ __forceinline__ void store_acc(float* g, v8f acc, int mt, int nt, int lane) {
    int col   = nt * 16 + (lane & 15);
    int mbase = mt * 16 + ((lane >> 4) << 3);
#pragma unroll
    for (int r = 0; r < 8; ++r)
        g[(mbase + r) * GSTR + col] = acc[r];
}

__global__ __launch_bounds__(NTHREADS) void lstm2_wmma_kernel(
    const float* __restrict__ x,
    const float* __restrict__ Wih1, const float* __restrict__ Whh1,
    const float* __restrict__ bih1, const float* __restrict__ bhh1,
    const float* __restrict__ Wih2, const float* __restrict__ Whh2,
    const float* __restrict__ bih2, const float* __restrict__ bhh2,
    const float* __restrict__ Wl,   const float* __restrict__ bl,
    float* __restrict__ out)
{
    __shared__ __align__(16) _Float16 h1s[BT * HSTR];
    __shared__ __align__(16) _Float16 h2s[BT * HSTR];
    __shared__ float gates[BT * GSTR];
    __shared__ float partials[BT * 65];
    __shared__ float b1p[GP], b2p[GP], wih1p[GP];
    __shared__ float wlp[HP];
    __shared__ float xbuf[BT];
    __shared__ float bls;

    const int tid  = threadIdx.x;
    const int lane = tid & 31;
    const int wave = tid >> 5;
    const int gb0  = blockIdx.x * BT;
    const int nt0  = wave * 2, nt1 = wave * 2 + 1;

    // --- one-time: weight B-operands resident in registers for the whole recurrence ---
    v16h W1[2][2], W2i[2][2], W2h[2][2];
#pragma unroll
    for (int t = 0; t < 2; ++t) {
        int nt = t ? nt1 : nt0;
#pragma unroll
        for (int kc = 0; kc < 2; ++kc) {
            W1[t][kc]  = make_B(Whh1, nt, kc, lane);
            W2i[t][kc] = make_B(Wih2, nt, kc, lane);
            W2h[t][kc] = make_B(Whh2, nt, kc, lane);
        }
    }

    // --- init LDS state ---
    for (int i = tid; i < BT * HSTR; i += NTHREADS) {
        h1s[i] = (_Float16)0.0f; h2s[i] = (_Float16)0.0f;
    }
    for (int i = tid; i < GP; i += NTHREADS) {
        int g = i >> 6, n = i & 63;
        float bb1 = 0.0f, bb2 = 0.0f, wi = 0.0f;
        if (n < H) {
            bb1 = bih1[g * H + n] + bhh1[g * H + n];
            bb2 = bih2[g * H + n] + bhh2[g * H + n];
            wi  = Wih1[g * H + n];          // Wih1 is [4H,1]
        }
        b1p[i] = bb1; b2p[i] = bb2; wih1p[i] = wi;
    }
    if (tid < HP) wlp[tid] = (tid < H) ? Wl[tid] : 0.0f;
    if (tid == 0) bls = bl[0];
    __syncthreads();

    float c1r[8], c2r[8];
#pragma unroll
    for (int r = 0; r < 8; ++r) { c1r[r] = 0.0f; c2r[r] = 0.0f; }

    for (int s = 0; s < OUTW; ++s) {
        // step input: while s<T stream x (uniform branch -> pure global load);
        // for s>=T, xbuf was already written by phase E of the previous step.
        if (s < TT) {
            if (tid < BT) xbuf[tid] = x[(size_t)(gb0 + tid) * TT + s];
        }

        // --- Phase A: gates1 = h1 @ Whh1p^T (K=64 -> 2 chained WMMAs) ---
#pragma unroll
        for (int mt = 0; mt < 2; ++mt) {
            v16h a0 = load_A(h1s, mt, 0, lane);
            v16h a1 = load_A(h1s, mt, 1, lane);
            v8f acc0 = {}; v8f acc1 = {};
            acc0 = wmma_f16(a0, W1[0][0], acc0); acc0 = wmma_f16(a1, W1[0][1], acc0);
            acc1 = wmma_f16(a0, W1[1][0], acc1); acc1 = wmma_f16(a1, W1[1][1], acc1);
            store_acc(gates, acc0, mt, nt0, lane);
            store_acc(gates, acc1, mt, nt1, lane);
        }
        __syncthreads();

        // --- Phase B: cell1 elementwise; c1 lives in registers ---
#pragma unroll
        for (int r = 0; r < 8; ++r) {
            int item = tid + NTHREADS * r;
            int b = item >> 6, n = item & 63;   // b = (tid>>6)+4r : fixed per (tid,r)
            float xb = xbuf[b];
            const float* gp = gates + b * GSTR;
            float gi = gp[n]       + wih1p[n]       * xb + b1p[n];
            float gf = gp[64 + n]  + wih1p[64 + n]  * xb + b1p[64 + n];
            float gg = gp[128 + n] + wih1p[128 + n] * xb + b1p[128 + n];
            float go = gp[192 + n] + wih1p[192 + n] * xb + b1p[192 + n];
            float c = sigm(gf) * c1r[r] + sigm(gi) * tanh_fast(gg);
            c1r[r] = c;
            h1s[b * HSTR + n] = (_Float16)(sigm(go) * tanh_fast(c));
        }
        __syncthreads();

        // --- Phase C: gates2 = h1new @ Wih2p^T + h2 @ Whh2p^T (4 chained WMMAs) ---
#pragma unroll
        for (int mt = 0; mt < 2; ++mt) {
            v16h a0 = load_A(h1s, mt, 0, lane);
            v16h a1 = load_A(h1s, mt, 1, lane);
            v16h h0 = load_A(h2s, mt, 0, lane);
            v16h h1v = load_A(h2s, mt, 1, lane);
            v8f acc0 = {}; v8f acc1 = {};
            acc0 = wmma_f16(a0, W2i[0][0], acc0); acc0 = wmma_f16(a1, W2i[0][1], acc0);
            acc0 = wmma_f16(h0, W2h[0][0], acc0); acc0 = wmma_f16(h1v, W2h[0][1], acc0);
            acc1 = wmma_f16(a0, W2i[1][0], acc1); acc1 = wmma_f16(a1, W2i[1][1], acc1);
            acc1 = wmma_f16(h0, W2h[1][0], acc1); acc1 = wmma_f16(h1v, W2h[1][1], acc1);
            store_acc(gates, acc0, mt, nt0, lane);
            store_acc(gates, acc1, mt, nt1, lane);
        }
        __syncthreads();

        // --- Phase D: cell2 elementwise; c2 in registers; emit y partials ---
#pragma unroll
        for (int r = 0; r < 8; ++r) {
            int item = tid + NTHREADS * r;
            int b = item >> 6, n = item & 63;
            const float* gp = gates + b * GSTR;
            float gi = gp[n]       + b2p[n];
            float gf = gp[64 + n]  + b2p[64 + n];
            float gg = gp[128 + n] + b2p[128 + n];
            float go = gp[192 + n] + b2p[192 + n];
            float c = sigm(gf) * c2r[r] + sigm(gi) * tanh_fast(gg);
            c2r[r] = c;
            float h2v = sigm(go) * tanh_fast(c);
            h2s[b * HSTR + n] = (_Float16)h2v;
            partials[b * 65 + n] = h2v * wlp[n];   // stride-65 pad: conflict-free reduce
        }
        __syncthreads();

        // --- Phase E: y = h2 @ Wl^T + bl ; write out; feed back as next x if needed ---
        if (tid < BT) {
            float y = bls;
#pragma unroll 8
            for (int n = 0; n < HP; ++n) y += partials[tid * 65 + n];
            out[(size_t)(gb0 + tid) * OUTW + s] = y;
            if (s + 1 >= TT) xbuf[tid] = y;   // next step consumes y as its input
        }
        __syncthreads();
    }
}

extern "C" void kernel_launch(void* const* d_in, const int* in_sizes, int n_in,
                              void* d_out, int out_size, void* d_ws, size_t ws_size,
                              hipStream_t stream) {
    const float* x    = (const float*)d_in[0];
    const float* Wih1 = (const float*)d_in[1];
    const float* Whh1 = (const float*)d_in[2];
    const float* bih1 = (const float*)d_in[3];
    const float* bhh1 = (const float*)d_in[4];
    const float* Wih2 = (const float*)d_in[5];
    const float* Whh2 = (const float*)d_in[6];
    const float* bih2 = (const float*)d_in[7];
    const float* bhh2 = (const float*)d_in[8];
    const float* Wl   = (const float*)d_in[9];
    const float* bl   = (const float*)d_in[10];
    float* out = (float*)d_out;

    dim3 grid(16384 / BT);   // 512 workgroups, one batch tile each
    dim3 block(NTHREADS);    // 8 wave32 waves
    lstm2_wmma_kernel<<<grid, block, 0, stream>>>(
        x, Wih1, Whh1, bih1, bhh1, Wih2, Whh2, bih2, bhh2, Wl, bl, out);
}
